// PointNet2_22600117911993
// MI455X (gfx1250) — matrix operand order repaired and checked by reading, not compiled, not run on hardware
//
#include <hip/hip_runtime.h>
#include <math.h>
#include <stdint.h>

typedef __bf16 bf16_t;
typedef __attribute__((ext_vector_type(16))) __bf16 v16bf;
typedef __attribute__((ext_vector_type(8)))  __bf16 v8bf;
typedef __attribute__((ext_vector_type(8)))  float  v8f;
typedef __attribute__((ext_vector_type(4))) unsigned int v4u;
typedef __attribute__((ext_vector_type(8))) int v8i;
typedef __attribute__((ext_vector_type(4))) int v4i;

#ifndef __has_builtin
#define __has_builtin(x) 0
#endif
#if __has_builtin(__builtin_amdgcn_tensor_load_to_lds)
#define HAVE_TDM 1
#else
#define HAVE_TDM 0
#endif
#if __has_builtin(__builtin_amdgcn_s_wait_tensorcnt)
#define WAIT_TENSORCNT0() __builtin_amdgcn_s_wait_tensorcnt(0)
#else
#define WAIT_TENSORCNT0()
#endif

// ---------------------------------------------------------------------------
// Address-space helpers: force global_load/global_store (avoid flat_*)
// ---------------------------------------------------------------------------
__device__ __forceinline__ v8bf gload8(const bf16_t* p) {
    typedef const __attribute__((address_space(1))) v8bf* gp;
    return *(gp)(uintptr_t)p;
}
__device__ __forceinline__ void gstoref(float* p, float v) {
    typedef __attribute__((address_space(1))) float* gp;
    *(gp)(uintptr_t)p = v;
}
__device__ __forceinline__ void gstoreb(bf16_t* p, bf16_t v) {
    typedef __attribute__((address_space(1))) bf16_t* gp;
    *(gp)(uintptr_t)p = v;
}
__device__ __forceinline__ bf16_t gloadb(const bf16_t* p) {
    typedef const __attribute__((address_space(1))) bf16_t* gp;
    return *(gp)(uintptr_t)p;
}

// ---------------------------------------------------------------------------
// WMMA fragment helper
// ---------------------------------------------------------------------------
__device__ __forceinline__ v8f wmma_acc(v8bf alo, v8bf ahi, v8bf blo, v8bf bhi, v8f acc) {
    v16bf A, Bv;
#pragma unroll
    for (int i = 0; i < 8; ++i) { A[i] = alo[i]; A[i + 8] = ahi[i]; Bv[i] = blo[i]; Bv[i + 8] = bhi[i]; }
    return __builtin_amdgcn_wmma_f32_16x16x32_bf16(false, A, false, Bv, (short)0, acc, false, false);
}

// ---------------------------------------------------------------------------
// Tensor Data Mover: 2D tile load (global bf16 -> LDS), D# per ISA 8.3/8.4.
// tile = rows x rowElems, row stride in tensor = rowStride (elements).
// LDS receives rows packed contiguously (stride = rowElems).
// ---------------------------------------------------------------------------
__device__ __forceinline__ void tdm_load_2d(bf16_t* ldsDst, const bf16_t* gsrc,
                                            int rowElems, int rows, int rowStride) {
#if HAVE_TDM
    unsigned long long ga = (unsigned long long)(uintptr_t)gsrc;
    unsigned int lds_off = (unsigned int)(uintptr_t)ldsDst; // addr[31:0] = LDS offset
    v4u g0;
    g0[0] = 1u;                                            // count=1, user mode
    g0[1] = lds_off;                                       // lds_addr
    g0[2] = (unsigned int)(ga & 0xFFFFFFFFu);              // global_addr[31:0]
    g0[3] = (unsigned int)((ga >> 32) & 0x1FFFFFFu) | (2u << 30); // addr[56:32] | type=2
    unsigned int td0 = (unsigned int)rowStride;            // tensor_dim0
    unsigned int td1 = 1u << 20;                           // tensor_dim1 (ample)
    v8i g1;
    g1[0] = (int)(1u << 16);                               // data_size=1 (2 bytes)
    g1[1] = (int)((td0 & 0xFFFFu) << 16);                  // tensor_dim0[15:0] @ bits63:48
    g1[2] = (int)(((td0 >> 16) & 0xFFFFu) | ((td1 & 0xFFFFu) << 16));
    g1[3] = (int)(((td1 >> 16) & 0xFFFFu) | (((unsigned)rowElems & 0xFFFFu) << 16)); // tile_dim0
    g1[4] = (int)((unsigned)rows & 0xFFFFu);               // tile_dim1 (tile_dim2=0)
    g1[5] = (int)(unsigned)rowStride;                      // tensor_dim0_stride[31:0]
    g1[6] = 0;
    g1[7] = 0;
    v4i z4 = {0, 0, 0, 0};
#if __clang_major__ >= 23
    v8i z8 = {0, 0, 0, 0, 0, 0, 0, 0};
    __builtin_amdgcn_tensor_load_to_lds(g0, g1, z4, z4, z8, 0);
#else
    __builtin_amdgcn_tensor_load_to_lds(g0, g1, z4, z4, 0);
#endif
#else
    (void)ldsDst; (void)gsrc; (void)rowElems; (void)rows; (void)rowStride;
#endif
}

// ---------------------------------------------------------------------------
// Small helper kernels
// ---------------------------------------------------------------------------

__global__ void split_xyz_kernel(const float* __restrict__ in, float* __restrict__ oxyz,
                                 float* __restrict__ opts, int total) {
    int i = blockIdx.x * blockDim.x + threadIdx.x;
    if (i >= total) return;
    oxyz[(size_t)i * 3 + 0] = in[(size_t)i * 4 + 0];
    oxyz[(size_t)i * 3 + 1] = in[(size_t)i * 4 + 1];
    oxyz[(size_t)i * 3 + 2] = in[(size_t)i * 4 + 2];
    opts[i] = in[(size_t)i * 4 + 3];
}

__global__ void prep_layer_kernel(const float* __restrict__ W, const float* __restrict__ bias,
                                  const float* __restrict__ g, const float* __restrict__ bt,
                                  bf16_t* __restrict__ W16, float* __restrict__ sv,
                                  float* __restrict__ tv, int cout, int cin, int kpad,
                                  int coutPad, int mode) {
    int r = blockIdx.x * blockDim.x + threadIdx.x;
    if (r >= coutPad) return;
    const float inv = 0.999995000037f; // 1/sqrt(1+1e-5)
    if (r < cout) {
        float s_, t_;
        if (mode == 0) { s_ = g[r] * inv; t_ = bias[r] * s_ + bt[r]; }
        else           { s_ = 1.0f;       t_ = bias[r]; }
        sv[r] = s_; tv[r] = t_;
        for (int c = 0; c < kpad; ++c)
            gstoreb(W16 + (size_t)r * kpad + c, (bf16_t)((c < cin) ? W[(size_t)r * cin + c] : 0.0f));
    } else {
        sv[r] = 0.0f; tv[r] = 0.0f;
        for (int c = 0; c < kpad; ++c) gstoreb(W16 + (size_t)r * kpad + c, (bf16_t)0.0f);
    }
}

// Farthest point sampling: one block per batch
__global__ void fps_kernel(const float* __restrict__ xyz, int* __restrict__ out,
                           int N, int npoint) {
    const int T = 256;
    __shared__ float sval[T];
    __shared__ int   sidx[T];
    int b = blockIdx.x, t = threadIdx.x;
    const float* base = xyz + (size_t)b * N * 3;
    float dist[64];
    for (int i = 0; i < 64; ++i) dist[i] = 1e10f;
    int far = 0;
    for (int it = 0; it < npoint; ++it) {
        if (t == 0) out[(size_t)b * npoint + it] = far;
        float fx = base[(size_t)far * 3 + 0];
        float fy = base[(size_t)far * 3 + 1];
        float fz = base[(size_t)far * 3 + 2];
        float best = -1.0f; int bi = 0; int c = 0;
        for (int n = t; n < N; n += T, ++c) {
            float dx = base[(size_t)n * 3 + 0] - fx;
            float dy = base[(size_t)n * 3 + 1] - fy;
            float dz = base[(size_t)n * 3 + 2] - fz;
            float d = dx * dx + dy * dy + dz * dz;
            float dd = fminf(dist[c], d);
            dist[c] = dd;
            if (dd > best) { best = dd; bi = n; }
        }
        sval[t] = best; sidx[t] = bi;
        __syncthreads();
        for (int off = T / 2; off > 0; off >>= 1) {
            if (t < off && sval[t + off] > sval[t]) { sval[t] = sval[t + off]; sidx[t] = sidx[t + off]; }
            __syncthreads();
        }
        far = sidx[0];
        __syncthreads();
    }
}

__global__ void gather_xyz_kernel(const float* __restrict__ xyz, const int* __restrict__ idx,
                                  float* __restrict__ nxyz, int N, int S, int total) {
    int i = blockIdx.x * blockDim.x + threadIdx.x;
    if (i >= total) return;
    int b = i / S, s = i - b * S;
    int p = idx[(size_t)b * S + s];
    nxyz[(size_t)i * 3 + 0] = xyz[((size_t)b * N + p) * 3 + 0];
    nxyz[(size_t)i * 3 + 1] = xyz[((size_t)b * N + p) * 3 + 1];
    nxyz[(size_t)i * 3 + 2] = xyz[((size_t)b * N + p) * 3 + 2];
}

// First K indices (ascending) within radius; pad with first hit
__global__ void ball_query_kernel(const float* __restrict__ xyz, const float* __restrict__ nxyz,
                                  int* __restrict__ gidx, int N, int S, int K, float r2, int total) {
    int i = blockIdx.x * blockDim.x + threadIdx.x;
    if (i >= total) return;
    int b = i / S;
    const float* base = xyz + (size_t)b * N * 3;
    float qx = nxyz[(size_t)i * 3 + 0];
    float qy = nxyz[(size_t)i * 3 + 1];
    float qz = nxyz[(size_t)i * 3 + 2];
    int* o = gidx + (size_t)i * K;
    int cnt = 0;
    for (int n = 0; n < N && cnt < K; ++n) {
        float dx = base[(size_t)n * 3 + 0] - qx;
        float dy = base[(size_t)n * 3 + 1] - qy;
        float dz = base[(size_t)n * 3 + 2] - qz;
        if (dx * dx + dy * dy + dz * dz <= r2) o[cnt++] = n;
    }
    int f = (cnt > 0) ? o[0] : 0;
    for (; cnt < K; ++cnt) o[cnt] = f;
}

// 3-NN + inverse-distance weights
__global__ void knn3_kernel(const float* __restrict__ xyz1, const float* __restrict__ xyz2,
                            int* __restrict__ kidx, float* __restrict__ kw,
                            int N1, int N2, int total) {
    int i = blockIdx.x * blockDim.x + threadIdx.x;
    if (i >= total) return;
    int b = i / N1;
    const float* base = xyz2 + (size_t)b * N2 * 3;
    float qx = xyz1[(size_t)i * 3 + 0];
    float qy = xyz1[(size_t)i * 3 + 1];
    float qz = xyz1[(size_t)i * 3 + 2];
    float d0 = 1e30f, d1 = 1e30f, d2 = 1e30f;
    int i0 = 0, i1 = 0, i2 = 0;
    for (int n = 0; n < N2; ++n) {
        float dx = base[(size_t)n * 3 + 0] - qx;
        float dy = base[(size_t)n * 3 + 1] - qy;
        float dz = base[(size_t)n * 3 + 2] - qz;
        float d = dx * dx + dy * dy + dz * dz;
        if (d < d0)      { d2 = d1; i2 = i1; d1 = d0; i1 = i0; d0 = d; i0 = n; }
        else if (d < d1) { d2 = d1; i2 = i1; d1 = d; i1 = n; }
        else if (d < d2) { d2 = d; i2 = n; }
    }
    float w0 = 1.0f / (d0 + 1e-8f), w1 = 1.0f / (d1 + 1e-8f), w2 = 1.0f / (d2 + 1e-8f);
    float ws = w0 + w1 + w2;
    kidx[(size_t)i * 3 + 0] = i0; kidx[(size_t)i * 3 + 1] = i1; kidx[(size_t)i * 3 + 2] = i2;
    kw[(size_t)i * 3 + 0] = w0 / ws; kw[(size_t)i * 3 + 1] = w1 / ws; kw[(size_t)i * 3 + 2] = w2 / ws;
}

// Build bf16 X = [points1 | interp(points2)] padded to kpad
__global__ void fp_gather_kernel(const float* __restrict__ p1, const float* __restrict__ p2,
                                 const int* __restrict__ kidx, const float* __restrict__ kw,
                                 bf16_t* __restrict__ X, int N1, int N2, int C1, int C2,
                                 int kpad, long total) {
    long e = (long)blockIdx.x * blockDim.x + threadIdx.x;
    if (e >= total) return;
    long r = e / kpad; int c = (int)(e - r * kpad);
    int b = (int)(r / N1), n1 = (int)(r - (long)b * N1);
    float v = 0.0f;
    if (c < C1) {
        v = p1[((size_t)b * N1 + n1) * C1 + c];
    } else if (c < C1 + C2) {
        int cc = c - C1;
        int j0 = kidx[r * 3 + 0], j1 = kidx[r * 3 + 1], j2 = kidx[r * 3 + 2];
        float w0 = kw[r * 3 + 0], w1 = kw[r * 3 + 1], w2 = kw[r * 3 + 2];
        const float* pb = p2 + (size_t)b * N2 * C2;
        v = w0 * pb[(size_t)j0 * C2 + cc] + w1 * pb[(size_t)j1 * C2 + cc] + w2 * pb[(size_t)j2 * C2 + cc];
    }
    gstoreb(X + e, (bf16_t)v);
}

// ---------------------------------------------------------------------------
// Fused SA scale: gather -> 3x (WMMA GEMM + BN + ReLU) in LDS -> max-pool
// ---------------------------------------------------------------------------
__global__ void sa_fused_kernel(const float* __restrict__ xyz, const float* __restrict__ pts,
                                const float* __restrict__ nxyz, const int* __restrict__ gidx,
                                const bf16_t* W0, const float* s0, const float* t0, int c0,
                                const bf16_t* W1, const float* s1, const float* t1, int c1,
                                const bf16_t* W2, const float* s2, const float* t2, int c2,
                                float* __restrict__ outPts, int N, int S, int K, int C,
                                int kpad0, int outCtot, int chanOff, int maxC) {
    extern __shared__ bf16_t smem[];
    bf16_t* bufX = smem;
    bf16_t* bufY = smem + (size_t)K * maxC;

    int g = blockIdx.x;
    int b = g / S, sq = g - b * S;
    const int* gi = gidx + (size_t)g * K;

    // Gather [pts | xyz - new_xyz | 0-pad] into LDS as bf16
    int totalG = K * kpad0;
    for (int e = threadIdx.x; e < totalG; e += blockDim.x) {
        int j = e / kpad0, c = e - j * kpad0;
        int p = gi[j];
        float v = 0.0f;
        if (c < C)          v = pts[((size_t)b * N + p) * C + c];
        else if (c < C + 3) v = xyz[((size_t)b * N + p) * 3 + (c - C)] -
                                nxyz[((size_t)b * S + sq) * 3 + (c - C)];
        bufX[e] = (bf16_t)v;
    }
    __syncthreads();

    const bf16_t* Wl[3] = {W0, W1, W2};
    const float*  Sl[3] = {s0, s1, s2};
    const float*  Tl[3] = {t0, t1, t2};
    int co[3] = {c0, c1, c2};

    bf16_t* cur = bufX;
    bf16_t* nxt = bufY;
    int cin = kpad0;
    int wave = threadIdx.x >> 5, lane = threadIdx.x & 31;
    int lhalf = lane >> 4, lcol = lane & 15;

    for (int l = 0; l < 3; ++l) {
        int cout = co[l];
        int mt = K >> 4, nt = cout >> 4;
        const bf16_t* Wp = Wl[l];
        for (int tI = wave; tI < mt * nt; tI += 8) {
            int tm = tI % mt, tn = tI / mt;
            v8f acc = {0.f, 0.f, 0.f, 0.f, 0.f, 0.f, 0.f, 0.f};
            const bf16_t* ap = cur + (size_t)(tm * 16 + lcol) * cin + lhalf * 8;   // LDS
            const bf16_t* bp = Wp  + (size_t)(tn * 16 + lcol) * cin + lhalf * 8;   // global
            for (int kb = 0; kb < cin; kb += 32) {
                v8bf alo = *(const v8bf*)(ap + kb);
                v8bf ahi = *(const v8bf*)(ap + kb + 16);
                acc = wmma_acc(alo, ahi, gload8(bp + kb), gload8(bp + kb + 16), acc);
            }
            int n = tn * 16 + lcol;
            float s_ = Sl[l][n], t_ = Tl[l][n];
#pragma unroll
            for (int i = 0; i < 8; ++i) {
                float y = fmaxf(acc[i] * s_ + t_, 0.0f);
                int m = tm * 16 + i + 8 * lhalf;
                nxt[(size_t)m * cout + n] = (bf16_t)y;
            }
        }
        __syncthreads();
        bf16_t* tmp = cur; cur = nxt; nxt = tmp;
        cin = cout;
    }

    // Max-pool over K neighbors
    for (int c = threadIdx.x; c < cin; c += blockDim.x) {
        float m = -3.4e38f;
        for (int j = 0; j < K; ++j) m = fmaxf(m, (float)cur[(size_t)j * cin + c]);
        gstoref(outPts + ((size_t)b * S + sq) * outCtot + chanOff + c, m);
    }
}

// ---------------------------------------------------------------------------
// Block-slab WMMA GEMM with TDM staging:
//   each block stages X[m0:m0+128, kc:kc+KC] into LDS via tensor_load_to_lds,
//   8 waves compute a 128 x Npad slab. Y = act((X @ W^T) * s + t).
// act: 0 = relu, 1 = sigmoid, 2 = none
// ---------------------------------------------------------------------------
__global__ void gemm_flat_kernel(const bf16_t* __restrict__ X, const bf16_t* __restrict__ W,
                                 const float* __restrict__ sv, const float* __restrict__ tv,
                                 float* __restrict__ outF, bf16_t* __restrict__ outB,
                                 int M, int Nreal, int Npad, int kpad, int act, int KC) {
    extern __shared__ bf16_t Xs[];       // 128 * KC bf16
    int wave = threadIdx.x >> 5, lane = threadIdx.x & 31;
    int lhalf = lane >> 4, lcol = lane & 15;
    int m0 = blockIdx.x * 128;
    int nt = Npad >> 4;
    int tiles = 8 * nt;                  // 8 row-tiles per slab

    v8f zero = {0.f, 0.f, 0.f, 0.f, 0.f, 0.f, 0.f, 0.f};
    v8f acc[8];
#pragma unroll
    for (int i = 0; i < 8; ++i) acc[i] = zero;

    for (int kc = 0; kc < kpad; kc += KC) {
        int chunk = (kpad - kc < KC) ? (kpad - kc) : KC;
#if HAVE_TDM
        if (threadIdx.x == 0)
            tdm_load_2d(Xs, X + (size_t)m0 * kpad + kc, chunk, 128, kpad);
        WAIT_TENSORCNT0();
#else
        for (int e = threadIdx.x; e < 128 * chunk; e += 256) {
            int r = e / chunk, c = e - r * chunk;
            Xs[e] = gloadb(X + (size_t)(m0 + r) * kpad + kc + c);
        }
#endif
        __syncthreads();
        int ti = 0;
        for (int tI = wave; tI < tiles; tI += 8, ++ti) {
            int tm = tI & 7, tn = tI >> 3;
            const bf16_t* ap = Xs + (size_t)(tm * 16 + lcol) * chunk + lhalf * 8;        // LDS
            const bf16_t* bp = W + (size_t)(tn * 16 + lcol) * kpad + kc + lhalf * 8;     // global
            v8f a = acc[ti];
            for (int kb = 0; kb < chunk; kb += 32) {
                __builtin_prefetch(bp + kb + 128, 0, 1);
                v8bf alo = *(const v8bf*)(ap + kb);
                v8bf ahi = *(const v8bf*)(ap + kb + 16);
                a = wmma_acc(alo, ahi, gload8(bp + kb), gload8(bp + kb + 16), a);
            }
            acc[ti] = a;
        }
        __syncthreads();
    }

    // Epilogue
    int ti = 0;
    for (int tI = wave; tI < tiles; tI += 8, ++ti) {
        int tm = tI & 7, tn = tI >> 3;
        int n = tn * 16 + lcol;
        float s_ = sv[n], t_ = tv[n];
#pragma unroll
        for (int i = 0; i < 8; ++i) {
            float y = acc[ti][i] * s_ + t_;
            if (act == 0)      y = fmaxf(y, 0.0f);
            else if (act == 1) y = 1.0f / (1.0f + __expf(-y));
            int m = m0 + tm * 16 + i + 8 * lhalf;
            if (outF && n < Nreal) gstoref(outF + (size_t)m * Nreal + n, y);
            if (outB)              gstoreb(outB + (size_t)m * Npad + n, (bf16_t)y);
        }
    }
}

// ---------------------------------------------------------------------------
// Host orchestration
// ---------------------------------------------------------------------------

static inline int cdiv_i(long a, int b) { return (int)((a + b - 1) / b); }
static inline int rup32(int x) { return (x + 31) & ~31; }
static inline int rup16(int x) { return (x + 15) & ~15; }

extern "C" void kernel_launch(void* const* d_in, const int* in_sizes, int n_in,
                              void* d_out, int out_size, void* d_ws, size_t ws_size,
                              hipStream_t stream) {
    (void)in_sizes; (void)n_in; (void)out_size; (void)ws_size;
    const int B = 2;
    const int lvlN[5] = {16384, 4096, 1024, 256, 64};
    const int lvlC[5] = {1, 320, 640, 768, 2048};
    const int sa_np[4] = {4096, 1024, 256, 64};
    const float sa_r[4][3] = {{0.2f,0.4f,0.6f},{0.4f,0.8f,1.6f},{0.8f,1.6f,3.2f},{1.6f,3.2f,6.4f}};
    const int sa_K[4][3] = {{32,64,64},{16,32,32},{16,32,64},{16,32,64}};
    const int sa_ch[4][3][3] = {
        {{32,32,64},{64,64,128},{64,96,128}},
        {{64,64,128},{128,128,256},{128,128,256}},
        {{128,128,256},{128,128,256},{128,256,256}},
        {{256,256,512},{256,256,512},{256,512,1024}}};
    const int fp_cin[4] = {2816, 768, 448, 129};
    const int fp_lvl[4] = {3, 2, 1, 0};

    // ---- workspace carve ----
    char* wsp = (char*)d_ws;
    auto alloc = [&](size_t bytes) -> void* {
        void* p = (void*)wsp;
        wsp += (bytes + 255) & ~(size_t)255;
        return p;
    };
    float* lx[5]; float* lp[5];
    for (int l = 0; l < 5; ++l) {
        lx[l] = (float*)alloc((size_t)B * lvlN[l] * 3 * 4);
        lp[l] = (float*)alloc((size_t)B * lvlN[l] * lvlC[l] * 4);
    }
    int*    fpsIdx = (int*)alloc((size_t)B * 4096 * 4);
    int*    gidx   = (int*)alloc((size_t)B * 4096 * 64 * 4);
    int*    knnIdx = (int*)alloc((size_t)B * 16384 * 3 * 4);
    float*  knnW   = (float*)alloc((size_t)B * 16384 * 3 * 4);
    bf16_t* Xfp    = (bf16_t*)alloc((size_t)5242880 * 2);
    bf16_t* Ybf    = (bf16_t*)alloc((size_t)32768 * 128 * 2);
    float*  featA  = (float*)alloc((size_t)B * 4096 * 128 * 4);
    float*  featB  = (float*)alloc((size_t)B * 4096 * 128 * 4);
    bf16_t* headX  = (bf16_t*)alloc((size_t)32768 * 128 * 2);
    bf16_t* headH  = (bf16_t*)alloc((size_t)32768 * 128 * 2);

    // ---- layer metadata + weight prep ----
    struct LM { const float *W, *b, *g, *bt; int cin, cout, kpad, coutPad, mode;
                bf16_t* W16; float* sv; float* tv; };
    LM lm[46];
    int pi = 1;
    int li = 0;
    for (int L = 0; L < 4; ++L)
        for (int s = 0; s < 3; ++s) {
            int cin = lvlC[L] + 3;
            for (int j = 0; j < 3; ++j) {
                int cout = sa_ch[L][s][j];
                lm[li] = {(const float*)d_in[pi], (const float*)d_in[pi+1],
                          (const float*)d_in[pi+2], (const float*)d_in[pi+3],
                          cin, cout, rup32(cin), rup16(cout), 0, nullptr, nullptr, nullptr};
                pi += 4; cin = cout; ++li;
            }
        }
    for (int f = 0; f < 4; ++f) {
        int cin = fp_cin[f];
        for (int j = 0; j < 2; ++j) {
            lm[li] = {(const float*)d_in[pi], (const float*)d_in[pi+1],
                      (const float*)d_in[pi+2], (const float*)d_in[pi+3],
                      cin, 128, rup32(cin), 128, 0, nullptr, nullptr, nullptr};
            pi += 4; cin = 128; ++li;
        }
    }
    lm[44] = {(const float*)d_in[pi], (const float*)d_in[pi+1],
              (const float*)d_in[pi+2], (const float*)d_in[pi+3],
              128, 128, 128, 128, 0, nullptr, nullptr, nullptr};
    pi += 4;
    lm[45] = {(const float*)d_in[pi], (const float*)d_in[pi+1], nullptr, nullptr,
              128, 13, 128, 16, 1, nullptr, nullptr, nullptr};

    for (int i = 0; i < 46; ++i) {
        lm[i].W16 = (bf16_t*)alloc((size_t)lm[i].coutPad * lm[i].kpad * 2);
        lm[i].sv  = (float*)alloc((size_t)lm[i].coutPad * 4);
        lm[i].tv  = (float*)alloc((size_t)lm[i].coutPad * 4);
        prep_layer_kernel<<<cdiv_i(lm[i].coutPad, 256), 256, 0, stream>>>(
            lm[i].W, lm[i].b, lm[i].g, lm[i].bt, lm[i].W16, lm[i].sv, lm[i].tv,
            lm[i].cout, lm[i].cin, lm[i].kpad, lm[i].coutPad, lm[i].mode);
    }

    // ---- split input ----
    const float* xyz_in = (const float*)d_in[0];
    split_xyz_kernel<<<cdiv_i((long)B * 16384, 256), 256, 0, stream>>>(xyz_in, lx[0], lp[0], B * 16384);

    // ---- SA levels ----
    for (int L = 0; L < 4; ++L) {
        int N = lvlN[L], S = sa_np[L];
        fps_kernel<<<B, 256, 0, stream>>>(lx[L], fpsIdx, N, S);
        gather_xyz_kernel<<<cdiv_i((long)B * S, 256), 256, 0, stream>>>(lx[L], fpsIdx, lx[L + 1], N, S, B * S);
        int chanOff = 0;
        for (int s = 0; s < 3; ++s) {
            int K = sa_K[L][s];
            float r = sa_r[L][s];
            ball_query_kernel<<<cdiv_i((long)B * S, 256), 256, 0, stream>>>(
                lx[L], lx[L + 1], gidx, N, S, K, r * r, B * S);
            int lid = L * 9 + s * 3;
            int c0 = lm[lid].cout, c1 = lm[lid + 1].cout, c2 = lm[lid + 2].cout;
            int kpad0 = lm[lid].kpad;
            int maxC = kpad0;
            if (c0 > maxC) maxC = c0;
            if (c1 > maxC) maxC = c1;
            if (c2 > maxC) maxC = c2;
            size_t lds = (size_t)2 * K * maxC * sizeof(bf16_t);
            sa_fused_kernel<<<B * S, 256, lds, stream>>>(
                lx[L], lp[L], lx[L + 1], gidx,
                lm[lid].W16,     lm[lid].sv,     lm[lid].tv,     c0,
                lm[lid + 1].W16, lm[lid + 1].sv, lm[lid + 1].tv, c1,
                lm[lid + 2].W16, lm[lid + 2].sv, lm[lid + 2].tv, c2,
                lp[L + 1], N, S, K, lvlC[L], kpad0, lvlC[L + 1], chanOff, maxC);
            chanOff += c2;
        }
    }

    // ---- FP levels ----
    float* outHead = (float*)d_out;                          // [B,16384,13]
    float* outFeat = (float*)d_out + (size_t)B * 16384 * 13; // [B,16384,128]
    const float* prevFeat = lp[4];
    int C2 = 2048;
    for (int f = 0; f < 4; ++f) {
        int lvl = fp_lvl[f];
        int N1 = lvlN[lvl], N2 = lvlN[lvl + 1], C1 = lvlC[lvl];
        int M = B * N1;
        knn3_kernel<<<cdiv_i((long)M, 256), 256, 0, stream>>>(lx[lvl], lx[lvl + 1], knnIdx, knnW, N1, N2, M);
        int lid = 36 + f * 2;
        int kpad = lm[lid].kpad;
        long totalX = (long)M * kpad;
        fp_gather_kernel<<<cdiv_i(totalX, 256), 256, 0, stream>>>(
            lp[lvl], prevFeat, knnIdx, knnW, Xfp, N1, N2, C1, C2, kpad, totalX);
        // layer 0 -> Ybf (bf16, relu)
        {
            int KC = (kpad < 512) ? kpad : 512;
            gemm_flat_kernel<<<M / 128, 256, (size_t)128 * KC * 2, stream>>>(
                Xfp, lm[lid].W16, lm[lid].sv, lm[lid].tv, nullptr, Ybf, M, 128, 128, kpad, 0, KC);
        }
        // layer 1 -> feat (fp32, relu); last level also feeds head (bf16)
        float* oF = (f == 0) ? featA : (f == 1) ? featB : (f == 2) ? featA : outFeat;
        bf16_t* oB = (f == 3) ? headX : nullptr;
        {
            gemm_flat_kernel<<<M / 128, 256, (size_t)128 * 128 * 2, stream>>>(
                Ybf, lm[lid + 1].W16, lm[lid + 1].sv, lm[lid + 1].tv, oF, oB, M, 128, 128, 128, 0, 128);
        }
        prevFeat = oF; C2 = 128;
    }

    // ---- head ----
    {
        int M = B * 16384;
        gemm_flat_kernel<<<M / 128, 256, (size_t)128 * 128 * 2, stream>>>(
            headX, lm[44].W16, lm[44].sv, lm[44].tv, nullptr, headH, M, 128, 128, 128, 0, 128);
        gemm_flat_kernel<<<M / 128, 256, (size_t)128 * 128 * 2, stream>>>(
            headH, lm[45].W16, lm[45].sv, lm[45].tv, outHead, nullptr, M, 13, 16, 128, 1, 128);
    }
}
